// ColbertLoss_37907381355015
// MI455X (gfx1250) — matrix operand order repaired and checked by reading, not compiled, not run on hardware
//
#include <hip/hip_runtime.h>

// ---- CDNA5 (gfx1250) wave32 WMMA types ----
typedef __attribute__((ext_vector_type(16))) __bf16 v16bf;
typedef __attribute__((ext_vector_type(8)))  __bf16 bf16x8;
typedef __attribute__((ext_vector_type(8)))  float  v8f;

#define B_ 128
#define N_ 32
#define S_ 512
#define D_ 128

// ---------------- f32 -> bf16 (round to nearest even) ----------------
__device__ __forceinline__ unsigned short f2bf(float f) {
    unsigned int u = __float_as_uint(f);
    return (unsigned short)((u + 0x7FFFu + ((u >> 16) & 1u)) >> 16);
}

__global__ void cvt_bf16_x4(const float4* __restrict__ in,
                            ushort4* __restrict__ out, int n4) {
    int i = blockIdx.x * blockDim.x + threadIdx.x;
    if (i < n4) {
        float4 v = in[i];
        ushort4 o;
        o.x = f2bf(v.x); o.y = f2bf(v.y); o.z = f2bf(v.z); o.w = f2bf(v.w);
        out[i] = o;
    }
}

// max(a,b) for finite inputs via v_med3: med3(a, b, +inf) == max(a, b).
// Compiler-known intrinsic -> WMMA->VALU hazard handling applies, and no
// canonicalize ops are inserted (unlike fmaxf).
__device__ __forceinline__ float vmax(float a, float b) {
    return __builtin_amdgcn_fmed3f(a, b, __int_as_float(0x7f800000));
}

// lane-xor shuffle via ds_swizzle (group-of-32 mode: and=0x1f, or=0, xor=M).
// xor masks 1/2/4/8 stay inside a 16-lane half; mask 16 swaps halves.
template <int M>
__device__ __forceinline__ float swz_xor(float v) {
    return __int_as_float(
        __builtin_amdgcn_ds_swizzle(__float_as_int(v), (M << 10) | 0x1f));
}

// ---------------- MaxSim scores kernel ----------------
// Grid (128, 64): blockIdx.x = c (doc), blockIdx.y -> query batches b0 = 2*y, b0+1.
// 256 threads = 8 waves. Each wave owns a 64-column S-slice of doc c and
// computes FOUR 16-row query M-tiles (2 batches x 2 tiles) from each B
// fragment: 4 WMMAs per 32-byte fragment, 4 independent accumulator chains.
__global__ void __launch_bounds__(256)
maxsim_scores(const __bf16* __restrict__ q,
              const __bf16* __restrict__ d,
              float* __restrict__ scores) {
    const int c    = blockIdx.x;
    const int b0   = blockIdx.y * 2;
    const int lane = threadIdx.x & 31;
    const int wave = threadIdx.x >> 5;
    const int r16  = lane & 15;   // A: row M; B: column N (doc row)
    const int half = lane >> 4;   // which K half this lane holds

    __shared__ float part[4][8][16];   // [b-tile*2 + mtile][wave][row in tile]

    // ---- A fragments (16x32 bf16 layout), tiles t = bt*2 + mt ----
    // lane holds row (b0+bt)*32 + mt*16 + r16; V0-3 = K k0..k0+7,
    // V4-7 = K k0+8..k0+15, with k0 = half*16.
    v16bf a[4][4];   // [tile][kk]
#pragma unroll
    for (int t = 0; t < 4; ++t) {
        const int bt = t >> 1, mt = t & 1;
        const __bf16* qrow = q + (size_t)((b0 + bt) * N_ + mt * 16 + r16) * D_;
#pragma unroll
        for (int kk = 0; kk < 4; ++kk) {
            const int off = kk * 32 + half * 16;
            bf16x8 lo = *(const bf16x8*)(qrow + off);
            bf16x8 hi = *(const bf16x8*)(qrow + off + 8);
#pragma unroll
            for (int i = 0; i < 8; ++i) { a[t][kk][i] = lo[i]; a[t][kk][8 + i] = hi[i]; }
        }
    }

    // Running max over s-tiles, elementwise on the C layout
    v8f rm[4];
#pragma unroll
    for (int t = 0; t < 4; ++t)
#pragma unroll
        for (int i = 0; i < 8; ++i) rm[t][i] = __int_as_float(0xff800000);

    // B fragment (32x16 bf16): lane = column n (doc row), contiguous K run
    // of 16 starting at half*16. Wave's doc rows: c*512 + wave*64 + st*16 + r16
    const __bf16* dwave = d + (size_t)(c * S_ + wave * 64 + r16) * D_ + half * 16;

#pragma unroll 1
    for (int st = 0; st < 4; ++st) {
        const __bf16* p = dwave + (size_t)st * 16 * D_;
        v16bf bb[4];
#pragma unroll
        for (int kk = 0; kk < 4; ++kk) bb[kk] = *(const v16bf*)(p + kk * 32);

        v8f acc[4];
#pragma unroll
        for (int t = 0; t < 4; ++t) acc[t] = (v8f){};

#pragma unroll
        for (int kk = 0; kk < 4; ++kk) {
#pragma unroll
            for (int t = 0; t < 4; ++t) {
                acc[t] = __builtin_amdgcn_wmma_f32_16x16x32_bf16(
                    false, a[t][kk], false, bb[kk], (short)0, acc[t], false, false);
            }
        }
#pragma unroll
        for (int t = 0; t < 4; ++t)
#pragma unroll
            for (int i = 0; i < 8; ++i) rm[t][i] = vmax(rm[t][i], acc[t][i]);
    }

    // Column-max within each 16-lane half (VGPR i holds rows i / i+8)
#pragma unroll
    for (int t = 0; t < 4; ++t) {
#pragma unroll
        for (int i = 0; i < 8; ++i) {
            float v = rm[t][i];
            v = vmax(v, swz_xor<1>(v));
            v = vmax(v, swz_xor<2>(v));
            v = vmax(v, swz_xor<4>(v));
            v = vmax(v, swz_xor<8>(v));
            rm[t][i] = v;
        }
    }
    if (r16 == 0) {
#pragma unroll
        for (int t = 0; t < 4; ++t)
#pragma unroll
            for (int i = 0; i < 8; ++i) part[t][wave][half * 8 + i] = rm[t][i];
    }
    __syncthreads();

    // Waves 0/1 finalize batches b0 / b0+1: combine 8 waves' maxima,
    // then sum the 32 per-query-row maxima of that batch.
    if (threadIdx.x < 64) {
        const int bt = threadIdx.x >> 5;    // which batch
        const int n  = threadIdx.x & 31;    // query row 0..31
        const int tt = bt * 2 + (n >> 4);   // tile index
        const int rr = n & 15;
        float v = part[tt][0][rr];
#pragma unroll
        for (int w = 1; w < 8; ++w) v = vmax(v, part[tt][w][rr]);
        v += swz_xor<16>(v);
        v += swz_xor<8>(v);
        v += swz_xor<4>(v);
        v += swz_xor<2>(v);
        v += swz_xor<1>(v);
        if (n == 0) scores[(b0 + bt) * B_ + c] = v;
    }
}

// ---------------- cross-entropy on scores (diagonal labels) ----------------
__global__ void ce_loss(const float* __restrict__ scores, float* __restrict__ out) {
    __shared__ float buf[B_];
    const int t = threadIdx.x;
    const float* row = scores + (size_t)t * B_;
    float mx = -3.4e38f;
    for (int j = 0; j < B_; ++j) mx = fmaxf(mx, row[j]);
    float se = 0.f;
    for (int j = 0; j < B_; ++j) se += expf(row[j] - mx);
    float lp = row[t] - mx - logf(se);
    buf[t] = -lp;
    __syncthreads();
    if (t == 0) {
        float s = 0.f;
        for (int j = 0; j < B_; ++j) s += buf[j];
        out[0] = s / (float)B_;
    }
}

extern "C" void kernel_launch(void* const* d_in, const int* in_sizes, int n_in,
                              void* d_out, int out_size, void* d_ws, size_t ws_size,
                              hipStream_t stream) {
    const float* qf = (const float*)d_in[0];   // [128,32,128] f32
    const float* df = (const float*)d_in[1];   // [128,512,128] f32

    char* ws = (char*)d_ws;
    const size_t qbytes = (size_t)B_ * N_ * D_ * 2;   // 1 MB bf16
    const size_t dbytes = (size_t)B_ * S_ * D_ * 2;   // 16 MB bf16
    __bf16* qbf  = (__bf16*)(ws);
    __bf16* dbf  = (__bf16*)(ws + qbytes);
    float* scores = (float*)(ws + qbytes + dbytes);   // 64 KB

    const int nq4 = (B_ * N_ * D_) / 4;
    const int nd4 = (B_ * S_ * D_) / 4;
    cvt_bf16_x4<<<(nq4 + 255) / 256, 256, 0, stream>>>((const float4*)qf, (ushort4*)qbf, nq4);
    cvt_bf16_x4<<<(nd4 + 255) / 256, 256, 0, stream>>>((const float4*)df, (ushort4*)dbf, nd4);

    dim3 grid(B_, B_ / 2);
    maxsim_scores<<<grid, 256, 0, stream>>>(qbf, dbf, scores);

    ce_loss<<<1, B_, 0, stream>>>(scores, (float*)d_out);
}
